// BoxHead_19696720020037
// MI455X (gfx1250) — matrix operand order repaired and checked by reading, not compiled, not run on hardware
//
#include <hip/hip_runtime.h>
#include <math.h>

typedef __attribute__((ext_vector_type(16))) __bf16 v16bf;
typedef __attribute__((ext_vector_type(8)))  __bf16 v8bf;
typedef __attribute__((ext_vector_type(8)))  float  v8f;

#define RCNT 1024   // total ROIs (B=2 * N=512)
#define CCH  256
#define FDIM 12544  // 256*7*7
#define HID  1024
#define KB   32
#define BSTRIDE 40  // LDS row stride in bf16 elems: 80B, keeps b128 reads aligned

// ---------------------------------------------------------------------------
// Kernel 1: multi-level ROI align -> fv [R, 12544] in bf16
// One block (256 threads) per ROI; bilinear coeffs precomputed in LDS.
// ---------------------------------------------------------------------------
__global__ __launch_bounds__(256) void roi_align_kernel(
    const float* __restrict__ p2, const float* __restrict__ p3,
    const float* __restrict__ p4, const float* __restrict__ p5,
    const float* __restrict__ proposals, __bf16* __restrict__ fv)
{
    const int r = blockIdx.x;
    const int b = r >> 9;                 // N = 512 proposals per batch

    const float x1 = proposals[r * 4 + 0];
    const float y1 = proposals[r * 4 + 1];
    const float x2 = proposals[r * 4 + 2];
    const float y2 = proposals[r * 4 + 3];

    // FPN level selection: clip(floor(4 + log(sqrt(w*h)/224)), 2, 5) - 2
    const float bw_ = x2 - x1, bh_ = y2 - y1;
    float lf = floorf(4.0f + logf(sqrtf(bw_ * bh_) / 224.0f));
    lf = fminf(fmaxf(lf, 2.0f), 5.0f);
    const int lvl = (int)lf - 2;

    const int   Hs[4] = {200, 100, 50, 25};
    const int   Ws[4] = {272, 136, 68, 34};
    const float* const feats[4] = {p2, p3, p4, p5};
    const int H = Hs[lvl], W = Ws[lvl];
    const float inv_stride = 1.0f / (float)(4 << lvl);   // strides 4,8,16,32
    const float* feat = feats[lvl] + (size_t)b * CCH * H * W;

    const float fx1 = x1 * inv_stride, fy1 = y1 * inv_stride;
    const float fx2 = x2 * inv_stride, fy2 = y2 * inv_stride;
    const float roi_w = fmaxf(fx2 - fx1, 1.0f);
    const float roi_h = fmaxf(fy2 - fy1, 1.0f);
    const float bw = roi_w / 7.0f, bh = roi_h / 7.0f;

    __shared__ int   xlo[14], xhi[14], ylo[14], yhi[14];
    __shared__ float wx0[14], wx1[14], wy0[14], wy1[14];

    const int t = threadIdx.x;
    if (t < 28) {
        const int  i   = t % 14;
        const bool isY = (t >= 14);
        const float g = (float)(i >> 1) + ((float)(i & 1) + 0.5f) * 0.5f;
        const float v = isY ? (fy1 + g * bh) : (fx1 + g * bw);
        const int   size = isY ? H : W;
        const bool  valid = (v >= -1.0f) && (v <= (float)size);
        const float vc = fminf(fmaxf(v, 0.0f), (float)size - 1.0f);
        const float lo = floorf(vc);
        const float frac = vc - lo;
        const int   lo_i = (int)lo;
        const int   hi_i = min(lo_i + 1, size - 1);
        const float w0 = valid ? (1.0f - frac) : 0.0f;
        const float w1 = valid ? frac : 0.0f;
        if (isY) { ylo[i] = lo_i; yhi[i] = hi_i; wy0[i] = w0; wy1[i] = w1; }
        else     { xlo[i] = lo_i; xhi[i] = hi_i; wx0[i] = w0; wx1[i] = w1; }
    }
    __syncthreads();

    const int HW = H * W;
    for (int o = t; o < FDIM; o += 256) {
        const int c   = o / 49;
        const int bin = o - c * 49;
        const int py  = bin / 7;
        const int px  = bin - py * 7;
        const float* fc = feat + (size_t)c * HW;
        float sum = 0.0f;
#pragma unroll
        for (int sy = 0; sy < 2; ++sy) {
            const int   iy = py * 2 + sy;
            const int   yA = ylo[iy], yB = yhi[iy];
            const float a0 = wy0[iy], a1 = wy1[iy];
#pragma unroll
            for (int sx = 0; sx < 2; ++sx) {
                const int   ix = px * 2 + sx;
                const int   xA = xlo[ix], xB = xhi[ix];
                const float c0 = wx0[ix], c1 = wx1[ix];
                sum += (fc[yA * W + xA] * c0 + fc[yA * W + xB] * c1) * a0
                     + (fc[yB * W + xA] * c0 + fc[yB * W + xB] * c1) * a1;
            }
        }
        fv[(size_t)r * FDIM + o] = (__bf16)(sum * 0.25f);
    }
}

// ---------------------------------------------------------------------------
// Kernel 2: C_bf16[M,N] = act(A_bf16[M,K] @ B_f32[K,N] + bias[N])
// Block tile 128x64, 8 waves, wave tile 32x32 -> 4 x v_wmma_f32_16x16x32_bf16
// per K-step. B tile staged in LDS as [n][k] bf16 so each B fragment is two
// aligned ds_load_b128; A fragments are two global_load_b128 per 16x32 frag.
// ---------------------------------------------------------------------------
__global__ __launch_bounds__(256) void gemm_bf16_wmma(
    const __bf16* __restrict__ A,   // [M,K] bf16 row-major
    const float*  __restrict__ B,   // [K,N] f32 row-major
    const float*  __restrict__ bias,
    __bf16*       __restrict__ C,   // [M,N] bf16
    int M, int N, int K, int do_relu)
{
    __shared__ __bf16 Bs[64 * BSTRIDE];   // [n][k], row stride 40 (80 B)

    const int tid  = threadIdx.x;
    const int lane = tid & 31;
    const int w    = tid >> 5;                    // wave 0..7
    const int m0   = blockIdx.y * 128 + (w & 3) * 32;
    const int nblk = blockIdx.x * 64;
    const int n0   = (w >> 2) * 32;               // wave's N offset inside tile

    v8f acc00 = {}, acc01 = {}, acc10 = {}, acc11 = {};

    // A-fragment lane mapping (16-bit A 16x32): row = m + (lane&15);
    // lanes 0-15 hold K = 0..7 & 16..23, lanes 16-31 hold K = 8..15 & 24..31
    const int arow0 = m0 + (lane & 15);
    const int kbase = (lane >> 4) * 8;
    // B-fragment lane mapping (B 32x16): N = lane&15; lanes 0-15 K = 0..7 & 16..23, lanes 16-31 K = 8..15 & 24..31
    const int kBrow = (lane >> 4) * 8;
    const int bn    = lane & 15;

    const __bf16* arow_p0 = A + (size_t)arow0 * K;        // rows m0..m0+15
    const __bf16* arow_p1 = arow_p0 + (size_t)16 * K;     // rows m0+16..m0+31
    const __bf16* bp0base = &Bs[(n0 + bn) * BSTRIDE + kBrow];
    const __bf16* bp1base = &Bs[(n0 + 16 + bn) * BSTRIDE + kBrow];

    for (int kk = 0; kk < K; kk += KB) {
        // Stage B tile (32 x 64 f32 -> bf16, transposed into [n][k])
        {
            const int r0 = tid >> 3;              // k row 0..31
            const int c0 = (tid & 7) * 8;         // n col 0,8,..,56
            const float* src = B + (size_t)(kk + r0) * N + nblk + c0;
            __builtin_prefetch(src + (size_t)KB * N, 0, 1);  // next K tile
#pragma unroll
            for (int j = 0; j < 8; ++j)
                Bs[(c0 + j) * BSTRIDE + r0] = (__bf16)src[j];
        }
        __syncthreads();

        // A fragments: two aligned 16-byte global loads each
        const v8bf* a0p = (const v8bf*)(arow_p0 + kk + kbase);
        const v8bf* a1p = (const v8bf*)(arow_p1 + kk + kbase);
        v16bf afr0 = __builtin_shufflevector(a0p[0], a0p[2],
                        0,1,2,3,4,5,6,7, 8,9,10,11,12,13,14,15);
        v16bf afr1 = __builtin_shufflevector(a1p[0], a1p[2],
                        0,1,2,3,4,5,6,7, 8,9,10,11,12,13,14,15);

        // B fragments: two aligned ds_load_b128 each (contiguous in k)
        const v8bf* b0p = (const v8bf*)bp0base;
        const v8bf* b1p = (const v8bf*)bp1base;
        v16bf bfr0 = __builtin_shufflevector(b0p[0], b0p[2],
                        0,1,2,3,4,5,6,7, 8,9,10,11,12,13,14,15);
        v16bf bfr1 = __builtin_shufflevector(b1p[0], b1p[2],
                        0,1,2,3,4,5,6,7, 8,9,10,11,12,13,14,15);

        acc00 = __builtin_amdgcn_wmma_f32_16x16x32_bf16(
                    false, afr0, false, bfr0, (short)0, acc00, false, false);
        acc01 = __builtin_amdgcn_wmma_f32_16x16x32_bf16(
                    false, afr0, false, bfr1, (short)0, acc01, false, false);
        acc10 = __builtin_amdgcn_wmma_f32_16x16x32_bf16(
                    false, afr1, false, bfr0, (short)0, acc10, false, false);
        acc11 = __builtin_amdgcn_wmma_f32_16x16x32_bf16(
                    false, afr1, false, bfr1, (short)0, acc11, false, false);
        __syncthreads();
    }

    // Epilogue: C/D layout -> row = mbase + v + (lane>>4)*8, col = lane&15
    const int col0  = nblk + n0 + (lane & 15);
    const int rb0   = m0 + (lane >> 4) * 8;        // rows for acc0x
    const int rb1   = rb0 + 16;                    // rows for acc1x
    const float bia0 = bias[col0];
    const float bia1 = bias[col0 + 16];
#pragma unroll
    for (int v = 0; v < 8; ++v) {
        float f00 = acc00[v] + bia0;
        float f01 = acc01[v] + bia1;
        float f10 = acc10[v] + bia0;
        float f11 = acc11[v] + bia1;
        if (do_relu) {
            f00 = fmaxf(f00, 0.0f); f01 = fmaxf(f01, 0.0f);
            f10 = fmaxf(f10, 0.0f); f11 = fmaxf(f11, 0.0f);
        }
        C[(size_t)(rb0 + v) * N + col0]      = (__bf16)f00;
        C[(size_t)(rb0 + v) * N + col0 + 16] = (__bf16)f01;
        C[(size_t)(rb1 + v) * N + col0]      = (__bf16)f10;
        C[(size_t)(rb1 + v) * N + col0 + 16] = (__bf16)f11;
    }
}

// ---------------------------------------------------------------------------
// Kernel 3: tiny heads. class_logits[r,4] then box_pred[r,12], concat in d_out.
// ---------------------------------------------------------------------------
__global__ __launch_bounds__(256) void head_kernel(
    const __bf16* __restrict__ X2,
    const float* __restrict__ Wc, const float* __restrict__ bc,
    const float* __restrict__ Wr, const float* __restrict__ br,
    float* __restrict__ out)
{
    const int idx = blockIdx.x * blockDim.x + threadIdx.x;
    if (idx >= RCNT * 16) return;
    const int r   = idx >> 4;
    const int col = idx & 15;
    const __bf16* x = X2 + (size_t)r * HID;
    float s = 0.0f;
    if (col < 4) {
        for (int k = 0; k < HID; ++k) s += (float)x[k] * Wc[k * 4 + col];
        out[r * 4 + col] = s + bc[col];
    } else {
        const int c = col - 4;
        for (int k = 0; k < HID; ++k) s += (float)x[k] * Wr[k * 12 + c];
        out[RCNT * 4 + r * 12 + c] = s + br[c];
    }
}

// ---------------------------------------------------------------------------
extern "C" void kernel_launch(void* const* d_in, const int* in_sizes, int n_in,
                              void* d_out, int out_size, void* d_ws, size_t ws_size,
                              hipStream_t stream) {
    const float* p2        = (const float*)d_in[0];
    const float* p3        = (const float*)d_in[1];
    const float* p4        = (const float*)d_in[2];
    const float* p5        = (const float*)d_in[3];
    const float* proposals = (const float*)d_in[4];
    const float* W1        = (const float*)d_in[5];
    const float* b1        = (const float*)d_in[6];
    const float* W2        = (const float*)d_in[7];
    const float* b2        = (const float*)d_in[8];
    const float* Wc        = (const float*)d_in[9];
    const float* bc        = (const float*)d_in[10];
    const float* Wr        = (const float*)d_in[11];
    const float* br        = (const float*)d_in[12];
    float* out = (float*)d_out;

    // Workspace layout (bf16 buffers)
    char* ws = (char*)d_ws;
    __bf16* fv = (__bf16*)ws;                                   // 1024*12544*2
    __bf16* X1 = (__bf16*)(ws + (size_t)RCNT * FDIM * 2);       // 1024*1024*2
    __bf16* X2 = (__bf16*)(ws + (size_t)RCNT * FDIM * 2 + (size_t)RCNT * HID * 2);

    // 1) ROI align -> fv (bf16)
    roi_align_kernel<<<RCNT, 256, 0, stream>>>(p2, p3, p4, p5, proposals, fv);

    // 2) X1 = relu(fv @ W1 + b1)   [1024 x 12544] x [12544 x 1024]
    gemm_bf16_wmma<<<dim3(HID / 64, RCNT / 128), 256, 0, stream>>>(
        fv, W1, b1, X1, RCNT, HID, FDIM, 1);

    // 3) X2 = relu(X1 @ W2 + b2)   [1024 x 1024] x [1024 x 1024]
    gemm_bf16_wmma<<<dim3(HID / 64, RCNT / 128), 256, 0, stream>>>(
        X1, W2, b2, X2, RCNT, HID, HID, 1);

    // 4) heads -> d_out = [class_logits (1024x4), box_pred (1024x12)]
    head_kernel<<<(RCNT * 16 + 255) / 256, 256, 0, stream>>>(X2, Wc, bc, Wr, br, out);
}
